// DTransformer_34540126994751
// MI455X (gfx1250) — matrix operand order repaired and checked by reading, not compiled, not run on hardware
//
#include <hip/hip_runtime.h>
#include <math.h>

// ---------------------------------------------------------------------------
// MI455X (gfx1250, wave32) transformer forward.
// Compute-bound (~760 GFLOP vs ~18us of HBM at 23.3 TB/s) -> everything heavy
// goes through v_wmma_f32_16x16x32_bf16. Global->LDS staging is two-phase and
// register-double-buffered across k-tiles so load latency hides under WMMAs.
// ---------------------------------------------------------------------------

#define L_MAX 2048
#define D_E   768
#define D_V   64
#define NHEAD 12
#define D_MLP 3072
#define VOCAB 32000
#define N_LAYERS 4

typedef __bf16 bf16x16 __attribute__((ext_vector_type(16)));
typedef __bf16 bf16x8  __attribute__((ext_vector_type(8)));
typedef float  f32x8   __attribute__((ext_vector_type(8)));
typedef unsigned short u16x8 __attribute__((ext_vector_type(8)));
typedef unsigned short u16;

// ---- helpers --------------------------------------------------------------

__device__ __forceinline__ u16 f2bf(float f) {
  unsigned int u = __float_as_uint(f);
  u += 0x7fffu + ((u >> 16) & 1u);          // round-to-nearest-even
  return (u16)(u >> 16);
}

__device__ __forceinline__ float gelu_tanh(float x) {
  float x3 = x * x * x;
  return 0.5f * x * (1.0f + tanhf(0.7978845608028654f * (x + 0.044715f * x3)));
}

// A-fragment for v_wmma_f32_16x16x32_bf16: lanes 0-15 read K[0..7]+K[16..23],
// lanes 16-31 read K[8..15]+K[24..31] of their row (caller bakes the +8 in).
__device__ __forceinline__ bf16x16 load_a_frag(const u16* p) {
  bf16x8 lo = *(const bf16x8*)(p);
  bf16x8 hi = *(const bf16x8*)(p + 16);
  bf16x16 r;
#pragma unroll
  for (int i = 0; i < 8; ++i) { r[i] = lo[i]; r[i + 8] = hi[i]; }
  return r;
}

__device__ __forceinline__ f32x8 wmma_bf16(bf16x16 a, bf16x16 b, f32x8 c) {
  return __builtin_amdgcn_wmma_f32_16x16x32_bf16(false, a, false, b,
                                                 (short)0, c, false, false);
}

__device__ __forceinline__ float half_reduce_max(float v) {
  v = fmaxf(v, __shfl_xor(v, 1, 32));
  v = fmaxf(v, __shfl_xor(v, 2, 32));
  v = fmaxf(v, __shfl_xor(v, 4, 32));
  v = fmaxf(v, __shfl_xor(v, 8, 32));
  return v;
}
__device__ __forceinline__ float half_reduce_sum(float v) {
  v += __shfl_xor(v, 1, 32);
  v += __shfl_xor(v, 2, 32);
  v += __shfl_xor(v, 4, 32);
  v += __shfl_xor(v, 8, 32);
  return v;
}

// ---- elementwise kernels --------------------------------------------------

__global__ __launch_bounds__(256) void embed_kernel(const int* __restrict__ x,
    const float* __restrict__ we, const float* __restrict__ pe,
    float* __restrict__ Y) {
  int row = blockIdx.x, tid = threadIdx.x;
  int id = x[row];
#pragma unroll
  for (int i = 0; i < 3; ++i) {
    int c = tid + i * 256;
    Y[(size_t)row * D_E + c] = we[(size_t)id * D_E + c] + pe[(size_t)row * D_E + c];
  }
}

__global__ __launch_bounds__(256) void zero_kernel(unsigned int* p, int n) {
  int i = blockIdx.x * 256 + threadIdx.x;
  if (i < n) p[i] = 0u;
}

// LayerNorm (population std, std==0 -> 1); bf16 out, optional (x + out) f32.
__global__ __launch_bounds__(256) void ln_kernel(const float* __restrict__ X,
    const float* __restrict__ g, const float* __restrict__ b,
    u16* __restrict__ outb, float* __restrict__ sumout) {
  __shared__ float red[256];
  int row = blockIdx.x, tid = threadIdx.x;
  const float* xr = X + (size_t)row * D_E;
  float x0 = xr[tid], x1 = xr[tid + 256], x2 = xr[tid + 512];

  red[tid] = x0 + x1 + x2;
  __syncthreads();
  for (int o = 128; o > 0; o >>= 1) { if (tid < o) red[tid] += red[tid + o]; __syncthreads(); }
  float mean = red[0] * (1.0f / D_E);
  __syncthreads();
  float d0 = x0 - mean, d1 = x1 - mean, d2 = x2 - mean;
  red[tid] = d0 * d0 + d1 * d1 + d2 * d2;
  __syncthreads();
  for (int o = 128; o > 0; o >>= 1) { if (tid < o) red[tid] += red[tid + o]; __syncthreads(); }
  float var = red[0] * (1.0f / D_E);
  float std = sqrtf(fmaxf(var, 0.0f));
  if (std == 0.0f) std = 1.0f;
  float inv = 1.0f / std;

  float xin[3] = {x0, x1, x2};
#pragma unroll
  for (int i = 0; i < 3; ++i) {
    int c = tid + i * 256;
    float n = g[c] * ((xin[i] - mean) * inv) + b[c];
    outb[(size_t)row * D_E + c] = f2bf(n);
    if (sumout) sumout[(size_t)row * D_E + c] = xin[i] + n;
  }
}

__global__ __launch_bounds__(256) void softmax_rows_kernel(float* __restrict__ X, int N) {
  __shared__ float red[256];
  int row = blockIdx.x, tid = threadIdx.x;
  float* xr = X + (size_t)row * N;
  float mx = -1e30f;
  for (int c = tid; c < N; c += 256) mx = fmaxf(mx, xr[c]);
  red[tid] = mx;
  __syncthreads();
  for (int o = 128; o > 0; o >>= 1) { if (tid < o) red[tid] = fmaxf(red[tid], red[tid + o]); __syncthreads(); }
  mx = red[0];
  __syncthreads();
  float s = 0.0f;
  for (int c = tid; c < N; c += 256) { float e = __expf(xr[c] - mx); xr[c] = e; s += e; }
  red[tid] = s;
  __syncthreads();
  for (int o = 128; o > 0; o >>= 1) { if (tid < o) red[tid] += red[tid + o]; __syncthreads(); }
  float invs = 1.0f / red[0];
  for (int c = tid; c < N; c += 256) xr[c] *= invs;
}

// ---- tiled WMMA GEMM ------------------------------------------------------
// C[M,N] = act(A_bf16[M,K] @ B_f32[K,N] + bias) + res_scale*res
// Tiles: 128x128x64, 8 waves -> 32x64 wave tiles, 16 wmma per k-tile/wave.
// k-tiles are register-double-buffered: next tile's global loads are issued
// before the current tile's WMMA block, so load latency hides under compute.
#define GA_STR 80   // 64 + 16 pad (ushorts); keeps 16B A-frag loads aligned
#define GB_STR 80   // keeps 16B B-frag loads aligned (80 % 8 == 0)

__global__ __launch_bounds__(256) void gemm_bf16_wmma(
    const u16* __restrict__ A, long long strideA,
    const float* __restrict__ B, long long strideB,
    const float* __restrict__ bias, long long strideBias,
    float* __restrict__ C32, u16* __restrict__ Cbf, long long strideC,
    int M, int N, int K, int lda, int ldb, int ldc,
    int act, const float* __restrict__ res, float res_scale) {
  __shared__ __align__(32) u16 sA[128 * GA_STR];
  __shared__ __align__(32) u16 sBt[128 * GB_STR];   // transposed: [n][k]

  const int bz = blockIdx.z;
  const u16*   Ab = A + (size_t)bz * strideA;
  const float* Bb = B + (size_t)bz * strideB;
  const float* biasb = bias ? (bias + (size_t)bz * strideBias) : (const float*)0;

  const int m0 = blockIdx.y * 128, n0 = blockIdx.x * 128;
  const int tid = threadIdx.x, wave = tid >> 5, lane = tid & 31;
  const int lh = lane >> 4, l15 = lane & 15;
  const int wm = wave >> 1, wn = wave & 1;

  // per-thread staging coordinates (constant over k-tiles)
  int arow[4], acol[4], brow[8], bcol[8];
#pragma unroll
  for (int i = 0; i < 4; ++i) { int c = tid + i * 256; arow[i] = c >> 3; acol[i] = (c & 7) * 8; }
#pragma unroll
  for (int i = 0; i < 8; ++i) { int c = tid + i * 256; brow[i] = c >> 5; bcol[i] = (c & 31) * 4; }

  f32x8 acc[2][4];
#pragma unroll
  for (int mi = 0; mi < 2; ++mi)
#pragma unroll
    for (int ni = 0; ni < 4; ++ni) { f32x8 z = {}; acc[mi][ni] = z; }

  const int ktiles = K >> 6;

  // prologue: issue all global loads for k-tile 0 (batched, no LDS yet)
  u16x8  aReg[4];
  float4 bReg[8];
#pragma unroll
  for (int i = 0; i < 4; ++i)
    aReg[i] = *(const u16x8*)(Ab + (size_t)(m0 + arow[i]) * lda + acol[i]);
#pragma unroll
  for (int i = 0; i < 8; ++i) {
    int gn = n0 + bcol[i];
    bReg[i] = (gn < N) ? *(const float4*)(Bb + (size_t)brow[i] * ldb + gn)
                       : make_float4(0.f, 0.f, 0.f, 0.f);
  }

  for (int kt = 0; kt < ktiles; ++kt) {
    // drain staged registers into LDS
#pragma unroll
    for (int i = 0; i < 4; ++i)
      *(u16x8*)(sA + arow[i] * GA_STR + acol[i]) = aReg[i];
#pragma unroll
    for (int i = 0; i < 8; ++i) {
      sBt[(bcol[i] + 0) * GB_STR + brow[i]] = f2bf(bReg[i].x);
      sBt[(bcol[i] + 1) * GB_STR + brow[i]] = f2bf(bReg[i].y);
      sBt[(bcol[i] + 2) * GB_STR + brow[i]] = f2bf(bReg[i].z);
      sBt[(bcol[i] + 3) * GB_STR + brow[i]] = f2bf(bReg[i].w);
    }
    __syncthreads();

    // issue next tile's global loads before computing (latency hides in WMMA)
    if (kt + 1 < ktiles) {
      const int k0n = (kt + 1) << 6;
#pragma unroll
      for (int i = 0; i < 4; ++i)
        aReg[i] = *(const u16x8*)(Ab + (size_t)(m0 + arow[i]) * lda + k0n + acol[i]);
#pragma unroll
      for (int i = 0; i < 8; ++i) {
        int gn = n0 + bcol[i];
        bReg[i] = (gn < N) ? *(const float4*)(Bb + (size_t)(k0n + brow[i]) * ldb + gn)
                           : make_float4(0.f, 0.f, 0.f, 0.f);
      }
      if (kt + 2 < ktiles)   // L2 warm-up two tiles ahead (global_prefetch)
        __builtin_prefetch(Bb + (size_t)(((kt + 2) << 6) + brow[0]) * ldb + n0 + bcol[0], 0, 1);
    }

    // compute current tile from LDS
#pragma unroll
    for (int ks = 0; ks < 2; ++ks) {
      bf16x16 af[2];
#pragma unroll
      for (int mi = 0; mi < 2; ++mi)
        af[mi] = load_a_frag(sA + (wm * 32 + mi * 16 + l15) * GA_STR + ks * 32 + lh * 8);
#pragma unroll
      for (int ni = 0; ni < 4; ++ni) {
        bf16x16 bf = load_a_frag(sBt + (wn * 64 + ni * 16 + l15) * GB_STR + ks * 32 + lh * 8);
#pragma unroll
        for (int mi = 0; mi < 2; ++mi)
          acc[mi][ni] = wmma_bf16(af[mi], bf, acc[mi][ni]);
      }
    }
    __syncthreads();
  }

  // epilogue: C layout -> VGPR r holds row (lh*8+r), column l15 of each subtile
#pragma unroll
  for (int mi = 0; mi < 2; ++mi)
#pragma unroll
    for (int ni = 0; ni < 4; ++ni)
#pragma unroll
      for (int r = 0; r < 8; ++r) {
        int row = m0 + wm * 32 + mi * 16 + lh * 8 + r;
        int col = n0 + wn * 64 + ni * 16 + l15;
        if (col < N) {
          float v = acc[mi][ni][r];
          if (biasb) v += biasb[col];
          if (act == 1) v = gelu_tanh(v);
          if (res) v += res_scale * res[(size_t)row * ldc + col];
          size_t off = (size_t)bz * strideC + (size_t)row * ldc + col;
          if (C32) C32[off] = v;
          if (Cbf) Cbf[off] = f2bf(v);
        }
      }
}

// ---- flash attention ------------------------------------------------------
// grid(L/128, H); 128 query rows / block, key blocks of 128, online softmax.
// q/k e-chunks are register-double-buffered; V-tile loads are issued before
// the softmax VALU block so their latency hides under the exp/reduce math.
// Writes the reference's overwrite semantics: head h<11 -> y[:,h] = O[:,0];
// head 11 -> y[:, 11:75] = O.
#define QK_STR 80    // 64 e-cols + pad
#define P_STR  136   // 128 keys + pad (A-frag 16B alignment: 136 % 8 == 0)
#define VT_STR 144   // 128 keys + pad (B-frag 16B alignment: 144 % 8 == 0)
#define SM_Q   0
#define SM_K   10240
#define SM_P   0
#define SM_VT  18432
#define SM_TOT 27648  // ushorts (54 KB)

__global__ __launch_bounds__(256) void attn_kernel(
    const u16* __restrict__ Q, const u16* __restrict__ K,
    const u16* __restrict__ V, u16* __restrict__ Yb, float scale) {
  __shared__ __align__(32) u16 smem[SM_TOT];
  u16* sQ = smem + SM_Q;
  u16* sK = smem + SM_K;
  u16* sP = smem + SM_P;
  u16* sV = smem + SM_VT;

  const int h = blockIdx.y, qb = blockIdx.x;
  const int tid = threadIdx.x, wave = tid >> 5, lane = tid & 31;
  const int lh = lane >> 4, l15 = lane & 15;

  const u16* Qh = Q + (size_t)h * L_MAX * D_E + (size_t)qb * 128 * D_E;
  const u16* Kh = K + (size_t)h * L_MAX * D_E;
  const u16* Vh = V + (size_t)h * L_MAX * D_V;

  // per-thread staging coordinates
  int srow[4], scol[4];
#pragma unroll
  for (int i = 0; i < 4; ++i) { int c = tid + i * 256; srow[i] = c >> 3; scol[i] = (c & 7) * 8; }

  float mrow[8], lrow[8];
  f32x8 O[4];
#pragma unroll
  for (int r = 0; r < 8; ++r) { mrow[r] = -1e30f; lrow[r] = 0.0f; }
#pragma unroll
  for (int ni = 0; ni < 4; ++ni) { f32x8 z = {}; O[ni] = z; }

  for (int kb = 0; kb < L_MAX / 128; ++kb) {
    const size_t kbase = (size_t)kb * 128;

    // ---- S = (q . k^T) over e in chunks of 64, pipelined ----
    f32x8 S[8];
#pragma unroll
    for (int ni = 0; ni < 8; ++ni) { f32x8 z = {}; S[ni] = z; }

    u16x8 qReg[4], kReg[4];
#pragma unroll
    for (int i = 0; i < 4; ++i) {
      qReg[i] = *(const u16x8*)(Qh + (size_t)srow[i] * D_E + scol[i]);
      kReg[i] = *(const u16x8*)(Kh + (kbase + srow[i]) * D_E + scol[i]);
    }

    for (int ec = 0; ec < D_E / 64; ++ec) {
#pragma unroll
      for (int i = 0; i < 4; ++i) {
        *(u16x8*)(sQ + srow[i] * QK_STR + scol[i]) = qReg[i];
        *(u16x8*)(sK + srow[i] * QK_STR + scol[i]) = kReg[i];
      }
      __syncthreads();

      if (ec + 1 < D_E / 64) {
        const int e0n = (ec + 1) * 64;
#pragma unroll
        for (int i = 0; i < 4; ++i) {
          qReg[i] = *(const u16x8*)(Qh + (size_t)srow[i] * D_E + e0n + scol[i]);
          kReg[i] = *(const u16x8*)(Kh + (kbase + srow[i]) * D_E + e0n + scol[i]);
        }
      }

      const int arow = wave * 16 + l15;
#pragma unroll
      for (int ks = 0; ks < 2; ++ks) {
        bf16x16 a = load_a_frag(sQ + arow * QK_STR + ks * 32 + lh * 8);
#pragma unroll
        for (int ni = 0; ni < 8; ++ni) {
          bf16x16 b = load_a_frag(sK + (ni * 16 + l15) * QK_STR + ks * 32 + lh * 8);
          S[ni] = wmma_bf16(a, b, S[ni]);
        }
      }
      __syncthreads();
    }

    // issue V-tile loads now; their latency hides under the softmax math below
    u16x8 vReg[4];
#pragma unroll
    for (int i = 0; i < 4; ++i)
      vReg[i] = *(const u16x8*)(Vh + (kbase + srow[i]) * D_V + scol[i]);

    // ---- online softmax (rows: lanes 0-15 -> r, lanes 16-31 -> 8+r) ----
#pragma unroll
    for (int r = 0; r < 8; ++r) {
      float rmax = -1e30f;
#pragma unroll
      for (int ni = 0; ni < 8; ++ni) { S[ni][r] *= scale; rmax = fmaxf(rmax, S[ni][r]); }
      rmax = half_reduce_max(rmax);
      float mn = fmaxf(mrow[r], rmax);
      float corr = __expf(mrow[r] - mn);
      float rsum = 0.0f;
#pragma unroll
      for (int ni = 0; ni < 8; ++ni) { float p = __expf(S[ni][r] - mn); S[ni][r] = p; rsum += p; }
      rsum = half_reduce_sum(rsum);
      lrow[r] = lrow[r] * corr + rsum;
      mrow[r] = mn;
#pragma unroll
      for (int ni = 0; ni < 4; ++ni) O[ni][r] *= corr;
    }

    // ---- spill P (bf16) to LDS, scatter V^T ----
    u16* Pw = sP + wave * 16 * P_STR;
#pragma unroll
    for (int ni = 0; ni < 8; ++ni)
#pragma unroll
      for (int r = 0; r < 8; ++r)
        Pw[(lh * 8 + r) * P_STR + ni * 16 + l15] = f2bf(S[ni][r]);
#pragma unroll
    for (int i = 0; i < 4; ++i)
#pragma unroll
      for (int j = 0; j < 8; ++j)
        sV[(scol[i] + j) * VT_STR + srow[i]] = vReg[i][j];
    __syncthreads();

    // ---- O += P @ V ----
#pragma unroll
    for (int ks = 0; ks < 4; ++ks) {
      bf16x16 a = load_a_frag(Pw + l15 * P_STR + ks * 32 + lh * 8);
#pragma unroll
      for (int ni = 0; ni < 4; ++ni) {
        bf16x16 b = load_a_frag(sV + (ni * 16 + l15) * VT_STR + ks * 32 + lh * 8);
        O[ni] = wmma_bf16(a, b, O[ni]);
      }
    }
    __syncthreads();
  }

  // ---- epilogue with reference's overwrite semantics ----
#pragma unroll
  for (int r = 0; r < 8; ++r) {
    float inv = 1.0f / lrow[r];
    int grow = qb * 128 + wave * 16 + lh * 8 + r;
#pragma unroll
    for (int ni = 0; ni < 4; ++ni) {
      float o = O[ni][r] * inv;
      int col = ni * 16 + l15;
      if (h < NHEAD - 1) {
        if (col == 0) Yb[(size_t)grow * D_E + h] = f2bf(o);
      } else {
        Yb[(size_t)grow * D_E + (NHEAD - 1) + col] = f2bf(o);
      }
    }
  }
}

// ---- host side ------------------------------------------------------------

extern "C" void kernel_launch(void* const* d_in, const int* in_sizes, int n_in,
                              void* d_out, int out_size, void* d_ws, size_t ws_size,
                              hipStream_t stream) {
  (void)in_sizes; (void)n_in; (void)out_size; (void)ws_size;
  const int*   x    = (const int*)d_in[0];
  const float* we   = (const float*)d_in[1];
  const float* pe   = (const float*)d_in[2];
  const float* Wq   = (const float*)d_in[3];
  const float* bq   = (const float*)d_in[4];
  const float* Wk   = (const float*)d_in[5];
  const float* bk   = (const float*)d_in[6];
  const float* Wv   = (const float*)d_in[7];
  const float* bv   = (const float*)d_in[8];
  const float* Wo   = (const float*)d_in[9];
  const float* bo   = (const float*)d_in[10];
  const float* g1   = (const float*)d_in[11];
  const float* be1  = (const float*)d_in[12];
  const float* g2   = (const float*)d_in[13];
  const float* be2  = (const float*)d_in[14];
  const float* W1   = (const float*)d_in[15];
  const float* bm1  = (const float*)d_in[16];
  const float* W2   = (const float*)d_in[17];
  const float* bm2  = (const float*)d_in[18];
  const float* gf   = (const float*)d_in[19];
  const float* bef  = (const float*)d_in[20];
  const float* Wu   = (const float*)d_in[21];
  const float* bu   = (const float*)d_in[22];
  float* out = (float*)d_out;

  // workspace carve-up (all sizes 256B-aligned)
  char* p = (char*)d_ws;
  float* Y  = (float*)p; p += (size_t)L_MAX * D_E * 4;          // residual stream
  float* R  = (float*)p; p += (size_t)L_MAX * D_E * 4;          // Y + LN(Y)
  u16*  XN  = (u16*)p;   p += (size_t)L_MAX * D_E * 2;          // LN out (bf16)
  u16*  Qb  = (u16*)p;   p += (size_t)NHEAD * L_MAX * D_E * 2;
  u16*  Kb  = (u16*)p;   p += (size_t)NHEAD * L_MAX * D_E * 2;
  u16*  Vb  = (u16*)p;   p += (size_t)NHEAD * L_MAX * D_V * 2;
  u16*  Ya  = (u16*)p;   p += (size_t)L_MAX * D_E * 2;          // attn concat (bf16)
  u16*  H1  = (u16*)p;   p += (size_t)L_MAX * D_MLP * 2;        // gelu(mlp1) bf16

  const float att_scale = 1.0f / sqrtf((float)D_E);

  embed_kernel<<<L_MAX, 256, 0, stream>>>(x, we, pe, Y);

  for (int i = 0; i < N_LAYERS; ++i) {
    const long long wqk_l = (long long)NHEAD * D_E * D_E;
    // xn = LN1(Y)
    ln_kernel<<<L_MAX, 256, 0, stream>>>(Y, g1 + i * D_E, be1 + i * D_E, XN, nullptr);
    // q,k,v projections (batched over heads)
    gemm_bf16_wmma<<<dim3(6, 16, NHEAD), 256, 0, stream>>>(
        XN, 0, Wq + (size_t)i * wqk_l, (long long)D_E * D_E,
        bq + (size_t)i * NHEAD * D_E, D_E,
        nullptr, Qb, (long long)L_MAX * D_E,
        L_MAX, D_E, D_E, D_E, D_E, D_E, 0, nullptr, 0.0f);
    gemm_bf16_wmma<<<dim3(6, 16, NHEAD), 256, 0, stream>>>(
        XN, 0, Wk + (size_t)i * wqk_l, (long long)D_E * D_E,
        bk + (size_t)i * NHEAD * D_E, D_E,
        nullptr, Kb, (long long)L_MAX * D_E,
        L_MAX, D_E, D_E, D_E, D_E, D_E, 0, nullptr, 0.0f);
    gemm_bf16_wmma<<<dim3(1, 16, NHEAD), 256, 0, stream>>>(
        XN, 0, Wv + (size_t)i * NHEAD * D_E * D_V, (long long)D_E * D_V,
        bv + (size_t)i * NHEAD * D_V, D_V,
        nullptr, Vb, (long long)L_MAX * D_V,
        L_MAX, D_V, D_E, D_E, D_V, D_V, 0, nullptr, 0.0f);
    // attention (writes sparse head-overwrite concat into Ya)
    zero_kernel<<<(L_MAX * D_E * 2 / 4 + 255) / 256, 256, 0, stream>>>(
        (unsigned int*)Ya, L_MAX * D_E * 2 / 4);
    attn_kernel<<<dim3(L_MAX / 128, NHEAD), 256, 0, stream>>>(Qb, Kb, Vb, Ya, att_scale);
    // Y = 2*Y + (Ya @ Wo + bo)
    gemm_bf16_wmma<<<dim3(6, 16, 1), 256, 0, stream>>>(
        Ya, 0, Wo + (size_t)i * D_E * D_E, 0, bo + (size_t)i * D_E, 0,
        Y, nullptr, 0,
        L_MAX, D_E, D_E, D_E, D_E, D_E, 0, Y, 2.0f);
    // zn = LN2(Y), R = Y + zn
    ln_kernel<<<L_MAX, 256, 0, stream>>>(Y, g2 + i * D_E, be2 + i * D_E, XN, R);
    // H1 = gelu(zn @ W1 + bm1)  (bf16)
    gemm_bf16_wmma<<<dim3(24, 16, 1), 256, 0, stream>>>(
        XN, 0, W1 + (size_t)i * D_E * D_MLP, 0, bm1 + (size_t)i * D_MLP, 0,
        nullptr, H1, 0,
        L_MAX, D_MLP, D_E, D_E, D_MLP, D_MLP, 1, nullptr, 0.0f);
    // Y = R + (H1 @ W2 + bm2)
    gemm_bf16_wmma<<<dim3(6, 16, 1), 256, 0, stream>>>(
        H1, 0, W2 + (size_t)i * D_MLP * D_E, 0, bm2 + (size_t)i * D_E, 0,
        Y, nullptr, 0,
        L_MAX, D_E, D_MLP, D_MLP, D_E, D_E, 0, R, 1.0f);
  }

  // final LN + unembed + softmax
  ln_kernel<<<L_MAX, 256, 0, stream>>>(Y, gf, bef, XN, nullptr);
  gemm_bf16_wmma<<<dim3(VOCAB / 128, 16, 1), 256, 0, stream>>>(
      XN, 0, Wu, 0, bu, 0,
      out, nullptr, 0,
      L_MAX, VOCAB, D_E, D_E, VOCAB, VOCAB, 0, nullptr, 0.0f);
  softmax_rows_kernel<<<L_MAX, 256, 0, stream>>>(out, VOCAB);
}
